// Self_Attention_New4_51977694216349
// MI455X (gfx1250) — compile-verified, matrix operand order checked by reading, not run on hardware
//
#include <hip/hip_runtime.h>
#include <hip/hip_bf16.h>

typedef __attribute__((ext_vector_type(16))) _Float16 v16h;
typedef __attribute__((ext_vector_type(8)))  _Float16 v8h;
typedef __attribute__((ext_vector_type(8)))  float    v8f;

// Problem constants (from reference setup_inputs)
#define BB 128
#define SS 200
#define HH 768
#define LL 256
#define NC 2

static constexpr size_t LH   = (size_t)LL * HH;        // 196608 elems per batch matrix
static constexpr size_t BLH  = (size_t)BB * LH;        // 25,165,824
static constexpr size_t SLL  = (size_t)LL * LL;        // 65536
static constexpr size_t SHH  = (size_t)HH * HH;        // 589824

// ---------------------------------------------------------------------------
// WMMA helpers
// ---------------------------------------------------------------------------
__device__ __forceinline__ v16h load_frag16(const _Float16* p) {
  // two contiguous 16B loads -> one 16-half fragment (per-lane K layout)
  v8h lo = *(const v8h*)p;
  v8h hi = *(const v8h*)(p + 16);
  return __builtin_shufflevector(lo, hi, 0,1,2,3,4,5,6,7,8,9,10,11,12,13,14,15);
}

__device__ __forceinline__ v8f wmma_f16(v16h a, v16h b, v8f c) {
  return __builtin_amdgcn_wmma_f32_16x16x32_f16(false, a, false, b, (short)0, c,
                                                false, false);
}

// ---------------------------------------------------------------------------
// Batched GEMM:  C[b][M,N] = A[b][M,K] * Bt[b][N,K]^T  (+bias, lrelu, gate)
// f16 in / f16 out, f32 accumulate.  grid = (N/256, M/128, batch), block = 256.
// 8 waves arranged 2x4; each wave computes a 64x64 C macro-tile as 4x4 WMMA
// 16x16x32 tiles (32 FLOP per operand byte; 1:1 wmma:b128 in the k-loop).
// Requires M%128==0, N%256==0, K%32==0 (all shapes here satisfy this).
// ---------------------------------------------------------------------------
__global__ void __launch_bounds__(256) gemm_bt(
    const _Float16* __restrict__ A,  unsigned long long strideA, int lda,
    const _Float16* __restrict__ Bt, unsigned long long strideB, int ldb,
    _Float16* __restrict__ C,        unsigned long long strideC, int ldc,
    const float* __restrict__ bias,  const _Float16* __restrict__ gate,
    int K, int applyLrelu)
{
  const int lane = threadIdx.x & 31;
  const int wave = threadIdx.x >> 5;
  const int lr   = lane & 15;
  const int hs   = lane >> 4;            // lane-half select
  const int m0   = blockIdx.y * 128 + (wave >> 2) * 64;
  const int n0   = blockIdx.x * 256 + (wave & 3) * 64;

  const _Float16* Ab = A  + (size_t)blockIdx.z * strideA;
  const _Float16* Bb = Bt + (size_t)blockIdx.z * strideB;

  const int koff = hs << 3;              // 0 or 8 (K sub-offset for this lane half)
  const _Float16* pa[4];
  const _Float16* pb[4];
#pragma unroll
  for (int i = 0; i < 4; ++i) {
    pa[i] = Ab + (size_t)(m0 + 16 * i + lr) * lda + koff;
    pb[i] = Bb + (size_t)(n0 + 16 * i + lr) * ldb + koff;
  }

  v8f acc[4][4] = {};
  for (int k0 = 0; k0 < K; k0 += 32) {
    v16h af[4], bf[4];
#pragma unroll
    for (int i = 0; i < 4; ++i) af[i] = load_frag16(pa[i] + k0);
#pragma unroll
    for (int j = 0; j < 4; ++j) bf[j] = load_frag16(pb[j] + k0);
#pragma unroll
    for (int i = 0; i < 4; ++i)
#pragma unroll
      for (int j = 0; j < 4; ++j)
        acc[i][j] = wmma_f16(af[i], bf[j], acc[i][j]);
  }

  _Float16* Cb = C + (size_t)blockIdx.z * strideC;
  const _Float16* Gb = gate ? gate + (size_t)blockIdx.z * strideC : nullptr;

#pragma unroll
  for (int i = 0; i < 4; ++i) {
    const int mbase = m0 + i * 16 + hs * 8;
#pragma unroll
    for (int j = 0; j < 4; ++j) {
      const int n = n0 + j * 16 + lr;
      const float bv = bias ? bias[n] : 0.0f;
#pragma unroll
      for (int r = 0; r < 8; ++r) {
        float v = acc[i][j][r] + bv;
        if (applyLrelu) v = (v > 0.0f) ? v : 0.01f * v;
        const size_t off = (size_t)(mbase + r) * ldc + n;
        if (Gb) v *= (float)Gb[off];
        Cb[off] = (_Float16)v;
      }
    }
  }
}

// ---------------------------------------------------------------------------
// Pad + convert tokens f32 [B,S,H] -> tp16 f16 [B,L,H] (rows S..L-1 zero)
// ---------------------------------------------------------------------------
__global__ void pad_convert(const float* __restrict__ tokens,
                            _Float16* __restrict__ tp16)
{
  size_t idx = (size_t)blockIdx.x * 256 + threadIdx.x;   // over B*L*H
  int h = (int)(idx % HH);
  size_t t = idx / HH;
  int l = (int)(t % LL);
  int b = (int)(t / LL);
  float v = (l < SS) ? tokens[((size_t)b * SS + l) * HH + h] : 0.0f;
  tp16[idx] = (_Float16)v;
}

// ---------------------------------------------------------------------------
// LDS-tiled transpose (+optional f32->f16 convert). in [R,Cc] -> out [Cc,R]
// R, Cc multiples of 32. block (32,8), grid (Cc/32, R/32, batch)
// ---------------------------------------------------------------------------
template <typename Tin>
__global__ void transpose_to_f16(const Tin* __restrict__ in,
                                 _Float16* __restrict__ out,
                                 int R, int Cc,
                                 unsigned long long inStride,
                                 unsigned long long outStride)
{
  __shared__ _Float16 tile[32][33];
  const Tin* inb  = in  + (size_t)blockIdx.z * inStride;
  _Float16*  outb = out + (size_t)blockIdx.z * outStride;
  const int c0 = blockIdx.x * 32, r0 = blockIdx.y * 32;
  const int tx = threadIdx.x, ty = threadIdx.y;
#pragma unroll
  for (int i = 0; i < 4; ++i) {
    int r = r0 + ty + i * 8, c = c0 + tx;
    tile[ty + i * 8][tx] = (_Float16)(float)inb[(size_t)r * Cc + c];
  }
  __syncthreads();
#pragma unroll
  for (int i = 0; i < 4; ++i) {
    int rr = c0 + ty + i * 8, cc = r0 + tx;
    outb[(size_t)rr * R + cc] = tile[tx][ty + i * 8];
  }
}

// ---------------------------------------------------------------------------
// Row softmax, in place on f16 rows. One wave32 per row; f32 math.
// ---------------------------------------------------------------------------
__global__ void softmax_rows(_Float16* __restrict__ S, int N, float scale,
                             int rowsTotal)
{
  int wave = blockIdx.x * (blockDim.x >> 5) + (threadIdx.x >> 5);
  if (wave >= rowsTotal) return;
  int lane = threadIdx.x & 31;
  _Float16* row = S + (size_t)wave * N;
  int cnt = N >> 5;                 // <= 24 for N<=768
  float vals[24];
  float mx = -3.0e38f;
  for (int i = 0; i < cnt; ++i) {
    float v = (float)row[lane + (i << 5)] * scale;
    vals[i] = v;
    mx = fmaxf(mx, v);
  }
  for (int off = 16; off > 0; off >>= 1) mx = fmaxf(mx, __shfl_xor(mx, off, 32));
  float sum = 0.0f;
  for (int i = 0; i < cnt; ++i) {
    float e = __expf(vals[i] - mx);
    vals[i] = e;
    sum += e;
  }
  for (int off = 16; off > 0; off >>= 1) sum += __shfl_xor(sum, off, 32);
  float inv = 1.0f / sum;
  for (int i = 0; i < cnt; ++i) row[lane + (i << 5)] = (_Float16)(vals[i] * inv);
}

// ---------------------------------------------------------------------------
// gate vector: g[r] = lrelu(dot(X[r,:K], w) + bias).  One wave per row.
// ---------------------------------------------------------------------------
__global__ void gate_vec(const _Float16* __restrict__ X,
                         const float* __restrict__ w,
                         const float* __restrict__ bias,
                         float* __restrict__ g, int K, int rowsTotal)
{
  int wave = blockIdx.x * (blockDim.x >> 5) + (threadIdx.x >> 5);
  if (wave >= rowsTotal) return;
  int lane = threadIdx.x & 31;
  const _Float16* row = X + (size_t)wave * K;
  float s = 0.0f;
  for (int i = lane; i < K; i += 32) s += (float)row[i] * w[i];
  for (int off = 16; off > 0; off >>= 1) s += __shfl_xor(s, off, 32);
  s += bias[0];
  s = (s > 0.0f) ? s : 0.01f * s;
  if (lane == 0) g[wave] = s;
}

// ---------------------------------------------------------------------------
// Gated elementwise multiply into [B,L,H]:
//  mode 0: out = grow[b*L+l]        * tp16     (TSGSA)
//  mode 1: out = gcol[b*H+h]        * tp16     (FSGSA)
//  mode 2: out = Gt[b][h*L+l]       * tp16     (FGSA, Gt is [H,L] f16)
// ---------------------------------------------------------------------------
__global__ void gated_mul(const _Float16* __restrict__ tp16,
                          const float* __restrict__ grow,
                          const float* __restrict__ gcol,
                          const _Float16* __restrict__ Gt,
                          _Float16* __restrict__ out, int mode)
{
  size_t idx = (size_t)blockIdx.x * 256 + threadIdx.x;
  int h = (int)(idx % HH);
  size_t t = idx / HH;
  int l = (int)(t % LL);
  int b = (int)(t / LL);
  float gv;
  if (mode == 0)      gv = grow[(size_t)b * LL + l];
  else if (mode == 1) gv = gcol[(size_t)b * HH + h];
  else                gv = (float)Gt[((size_t)b * HH + h) * LL + l];
  out[idx] = (_Float16)(gv * (float)tp16[idx]);
}

// ---------------------------------------------------------------------------
// Mean pool over L into pooled f16 [B, 4H], concat order o1|o2|o3|tp
// ---------------------------------------------------------------------------
__global__ void mean_pool(const _Float16* __restrict__ o1,
                          const _Float16* __restrict__ o2,
                          const _Float16* __restrict__ o3,
                          const _Float16* __restrict__ tp16,
                          _Float16* __restrict__ pooled)
{
  int idx = blockIdx.x * 256 + threadIdx.x;   // over B*4H
  int j = idx % (4 * HH);
  int b = idx / (4 * HH);
  int br = j / HH, h = j % HH;
  const _Float16* src = (br == 0) ? o1 : (br == 1) ? o2 : (br == 2) ? o3 : tp16;
  const _Float16* p = src + (size_t)b * LH + h;
  float s = 0.0f;
  for (int l = 0; l < LL; ++l) s += (float)p[(size_t)l * HH];
  pooled[idx] = (_Float16)(s * (1.0f / LL));
}

// ---------------------------------------------------------------------------
// Final projection: out[b,c] = dot(hidden[b,:H], Wf2[:,c]) + bf2[c]  (f32 out)
// One wave per (b,c).
// ---------------------------------------------------------------------------
__global__ void final_proj(const _Float16* __restrict__ hidden,
                           const float* __restrict__ Wf2,
                           const float* __restrict__ bf2,
                           float* __restrict__ out)
{
  int wave = blockIdx.x * (blockDim.x >> 5) + (threadIdx.x >> 5);
  if (wave >= BB * NC) return;
  int c = wave & (NC - 1), b = wave / NC;
  int lane = threadIdx.x & 31;
  float s = 0.0f;
  for (int h = lane; h < HH; h += 32)
    s += (float)hidden[(size_t)b * HH + h] * Wf2[h * NC + c];
  for (int off = 16; off > 0; off >>= 1) s += __shfl_xor(s, off, 32);
  if (lane == 0) out[b * NC + c] = s + bf2[c];
}

// ---------------------------------------------------------------------------
// Host orchestration
// ---------------------------------------------------------------------------
extern "C" void kernel_launch(void* const* d_in, const int* in_sizes, int n_in,
                              void* d_out, int out_size, void* d_ws, size_t ws_size,
                              hipStream_t stream)
{
  const float* tokens = (const float*)d_in[0];
  const float* Wq  = (const float*)d_in[1];  const float* bq  = (const float*)d_in[2];
  const float* Wk  = (const float*)d_in[3];  const float* bk  = (const float*)d_in[4];
  const float* Wv  = (const float*)d_in[5];  const float* bv  = (const float*)d_in[6];
  const float* Wfq = (const float*)d_in[7];  const float* bfq = (const float*)d_in[8];
  const float* Wfk = (const float*)d_in[9];  const float* bfk = (const float*)d_in[10];
  const float* Wfv = (const float*)d_in[11]; const float* bfv = (const float*)d_in[12];
  const float* Wtsg= (const float*)d_in[13]; const float* btsg= (const float*)d_in[14];
  const float* Wfsg= (const float*)d_in[15]; const float* bfsg= (const float*)d_in[16];
  const float* Wtg = (const float*)d_in[17]; const float* btg = (const float*)d_in[18];
  const float* Wfg = (const float*)d_in[19]; const float* bfg = (const float*)d_in[20];
  const float* Wf1 = (const float*)d_in[21]; const float* bf1 = (const float*)d_in[22];
  const float* Wf2 = (const float*)d_in[23]; const float* bf2 = (const float*)d_in[24];
  float* outp = (float*)d_out;

  // workspace carving (256B aligned)
  char* p = (char*)d_ws;
  auto alloc = [&](size_t elems, size_t esz) -> void* {
    void* r = (void*)p;
    p += ((elems * esz + 255) & ~(size_t)255);
    return r;
  };
  _Float16* tp16   = (_Float16*)alloc(BLH, 2);
  _Float16* tpT16  = (_Float16*)alloc(BLH, 2);
  _Float16* bufQ   = (_Float16*)alloc(BLH, 2);
  _Float16* bufK   = (_Float16*)alloc(BLH, 2);
  _Float16* bufV   = (_Float16*)alloc(BLH, 2);
  _Float16* bufT   = (_Float16*)alloc(BLH, 2);
  _Float16* tTSA   = (_Float16*)alloc(BLH, 2);
  _Float16* tFSA   = (_Float16*)alloc(BLH, 2);
  _Float16* tTGSA  = (_Float16*)alloc(BLH, 2);
  _Float16* tFGSA  = (_Float16*)alloc(BLH, 2);
  _Float16* tTSGSA = (_Float16*)alloc(BLH, 2);
  _Float16* tFSGSA = (_Float16*)alloc(BLH, 2);
  _Float16* out1   = (_Float16*)alloc(BLH, 2);
  _Float16* out2   = (_Float16*)alloc(BLH, 2);
  _Float16* out3   = (_Float16*)alloc(BLH, 2);
  _Float16* bufS   = (_Float16*)alloc((size_t)BB * SLL, 2);   // [B,L,L]
  _Float16* bufSN  = (_Float16*)alloc((size_t)BB * SHH, 2);   // [B,H,H]
  _Float16* wqT    = (_Float16*)alloc(SHH, 2);
  _Float16* wkT    = (_Float16*)alloc(SHH, 2);
  _Float16* wvT    = (_Float16*)alloc(SHH, 2);
  _Float16* wtgT   = (_Float16*)alloc(SHH, 2);
  _Float16* wfqT   = (_Float16*)alloc(SLL, 2);
  _Float16* wfkT   = (_Float16*)alloc(SLL, 2);
  _Float16* wfvT   = (_Float16*)alloc(SLL, 2);
  _Float16* wfgT   = (_Float16*)alloc(SLL, 2);
  _Float16* wf1T   = (_Float16*)alloc((size_t)4 * HH * HH, 2);
  _Float16* pooled = (_Float16*)alloc((size_t)BB * 4 * HH, 2);
  _Float16* hidden = (_Float16*)alloc((size_t)BB * HH, 2);
  float*    grow   = (float*)alloc((size_t)BB * LL, 4);
  float*    gcol   = (float*)alloc((size_t)BB * HH, 4);

  const float sh = 1.0f / sqrtf((float)HH);
  const float sl = 1.0f / sqrtf((float)LL);

  auto gemm = [&](const _Float16* A, unsigned long long sA, int lda,
                  const _Float16* Bt, unsigned long long sB, int ldb,
                  _Float16* C, unsigned long long sC, int ldc,
                  const float* bias, const _Float16* gate,
                  int M, int N, int K, int lrelu, int batch) {
    dim3 grid(N / 256, M / 128, batch);
    gemm_bt<<<grid, 256, 0, stream>>>(A, sA, lda, Bt, sB, ldb, C, sC, ldc,
                                      bias, gate, K, lrelu);
  };
  auto trf16 = [&](const _Float16* in, _Float16* out, int R, int Cc, int batch,
                   unsigned long long str) {
    dim3 grid(Cc / 32, R / 32, batch);
    transpose_to_f16<_Float16><<<grid, dim3(32, 8), 0, stream>>>(in, out, R, Cc, str, str);
  };
  auto trf32 = [&](const float* in, _Float16* out, int R, int Cc) {
    dim3 grid(Cc / 32, R / 32, 1);
    transpose_to_f16<float><<<grid, dim3(32, 8), 0, stream>>>(in, out, R, Cc, 0, 0);
  };
  auto smax = [&](_Float16* S, int N, float scale, int rows) {
    softmax_rows<<<rows / 8, 256, 0, stream>>>(S, N, scale, rows);
  };

  // 0) convert/pad activations; transpose+convert weights
  pad_convert<<<(unsigned)(BLH / 256), 256, 0, stream>>>(tokens, tp16);
  trf16(tp16, tpT16, LL, HH, BB, LH);
  trf32(Wq,  wqT,  HH, HH);  trf32(Wk,  wkT,  HH, HH);
  trf32(Wv,  wvT,  HH, HH);  trf32(Wtg, wtgT, HH, HH);
  trf32(Wfq, wfqT, LL, LL);  trf32(Wfk, wfkT, LL, LL);
  trf32(Wfv, wfvT, LL, LL);  trf32(Wfg, wfgT, LL, LL);
  trf32(Wf1, wf1T, 4 * HH, HH);

  // 1) token self-attention
  gemm(tp16, LH, HH, wqT, 0, HH, bufQ, LH, HH, bq, nullptr, LL, HH, HH, 0, BB);
  gemm(tp16, LH, HH, wkT, 0, HH, bufK, LH, HH, bk, nullptr, LL, HH, HH, 0, BB);
  gemm(tp16, LH, HH, wvT, 0, HH, bufV, LH, HH, bv, nullptr, LL, HH, HH, 0, BB);
  gemm(bufQ, LH, HH, bufK, LH, HH, bufS, SLL, LL, nullptr, nullptr, LL, LL, HH, 0, BB);
  smax(bufS, LL, sh, BB * LL);
  trf16(bufV, bufT, LL, HH, BB, LH);                         // V^T [H,L]
  gemm(bufS, SLL, LL, bufT, LH, LL, tTSA, LH, HH, nullptr, nullptr, LL, HH, LL, 0, BB);

  // 2) feature self-attention (reuse Q/K/V buffers)
  gemm(tpT16, LH, LL, wfqT, 0, LL, bufQ, LH, LL, bfq, nullptr, HH, LL, LL, 0, BB);
  gemm(tpT16, LH, LL, wfkT, 0, LL, bufK, LH, LL, bfk, nullptr, HH, LL, LL, 0, BB);
  gemm(tpT16, LH, LL, wfvT, 0, LL, bufV, LH, LL, bfv, nullptr, HH, LL, LL, 0, BB);
  gemm(bufQ, LH, LL, bufK, LH, LL, bufSN, SHH, HH, nullptr, nullptr, HH, HH, LL, 0, BB);
  smax(bufSN, HH, sl, BB * HH);
  trf16(bufV, bufT, HH, LL, BB, LH);                         // VN^T [L,H]
  gemm(bufSN, SHH, HH, bufT, LH, HH, bufQ, LH, LL, nullptr, nullptr, HH, LL, HH, 0, BB);
  trf16(bufQ, tFSA, HH, LL, BB, LH);                         // FSA [L,H]

  // 3) gated branches
  gemm(tp16, LH, HH, wtgT, 0, HH, tTGSA, LH, HH, btg, tp16, LL, HH, HH, 1, BB);
  gemm(tpT16, LH, LL, wfgT, 0, LL, bufK, LH, LL, bfg, nullptr, HH, LL, LL, 1, BB); // G [H,L]
  gated_mul<<<(unsigned)(BLH / 256), 256, 0, stream>>>(tp16, nullptr, nullptr, bufK, tFGSA, 2);
  gate_vec<<<(BB * LL) / 8, 256, 0, stream>>>(tp16, Wtsg, btsg, grow, HH, BB * LL);
  gated_mul<<<(unsigned)(BLH / 256), 256, 0, stream>>>(tp16, grow, nullptr, nullptr, tTSGSA, 0);
  gate_vec<<<(BB * HH) / 8, 256, 0, stream>>>(tpT16, Wfsg, bfsg, gcol, LL, BB * HH);
  gated_mul<<<(unsigned)(BLH / 256), 256, 0, stream>>>(tp16, nullptr, gcol, nullptr, tFSGSA, 1);

  // 4) three cross-attentions (att = softmax(a b^T / sqrt(H)); out = att @ tp)
  const _Float16* ca[3] = {tTSA, tTGSA, tTSGSA};
  const _Float16* cb[3] = {tFSA, tFGSA, tFSGSA};
  _Float16* co[3] = {out1, out2, out3};
  for (int i = 0; i < 3; ++i) {
    gemm(ca[i], LH, HH, cb[i], LH, HH, bufS, SLL, LL, nullptr, nullptr, LL, LL, HH, 0, BB);
    smax(bufS, LL, sh, BB * LL);
    gemm(bufS, SLL, LL, tpT16, LH, LL, co[i], LH, HH, nullptr, nullptr, LL, HH, LL, 0, BB);
  }

  // 5) pool + classifier head
  mean_pool<<<(BB * 4 * HH) / 256, 256, 0, stream>>>(out1, out2, out3, tp16, pooled);
  gemm(pooled, 0, 4 * HH, wf1T, 0, 4 * HH, hidden, 0, HH, bf1, nullptr,
       BB, HH, 4 * HH, 0, 1);
  final_proj<<<(BB * NC) / 8, 256, 0, stream>>>(hidden, Wf2, bf2, outp);

  (void)in_sizes; (void)n_in; (void)out_size; (void)ws_size;
}
